// RadianceFieldTT_23613730193525
// MI455X (gfx1250) — matrix operand order repaired and checked by reading, not compiled, not run on hardware
//
#include <hip/hip_runtime.h>

// ---------------------------------------------------------------------------
// QTT radiance-field sampler for gfx1250 (MI455X).
//
// Per level l the data-dependent matvec v' = v @ cores[l][:, d, :] is a dense
// GEMM v' = A @ Bflat with Bflat = cores[l] viewed as (512,64) (k = r*8+d) and
// A(p, r*8+d) = v(p,r) iff d == digit_p.  digit_p is constant per lane in the
// WMMA A layout, so A operands are built in registers with cndmask selects.
// v_wmma_f32_16x16x32_f16, f32 accumulate.
//
// B matrices stream through LDS, double-buffered with CDNA5 async DMA
// (global_load_async_to_lds_b128 / s_wait_asynccnt): a pre-pass kernel writes
// the f16, transposed, stride-padded LDS image of all 7 levels into d_ws, and
// the main kernel overlaps the level-(l+1) fill with the level-l WMMAs.
// Falls back to synchronous in-kernel conversion when d_ws is too small.
//
// LDS layout places the per-wave v-buffers at offset 0 so all scalar v reads
// in the inner loop use DS immediate offsets (no address VALU).
// ---------------------------------------------------------------------------

typedef __attribute__((ext_vector_type(16))) _Float16     v16h;
typedef __attribute__((ext_vector_type(8)))  float        v8f;
typedef __attribute__((ext_vector_type(4)))  unsigned int v4u;

#define NB    1024
#define NR    256
#define NTOT  (NB * NR)          // 262144 points
#define RANK  64
#define KDIM  512                // RANK * 8 digits
#define PAY   28                 // 27 SH coeffs + sigma
#define NLEV  8

#define WAVES   8                // 256 threads
#define PTS_W   16               // points per wave (one WMMA M-tile)
#define PTS_WG  (WAVES * PTS_W)  // 128 points per workgroup

// sBT: cores[l] transposed to s-major f16: sBT[s][k].  Row stride 520 f16 =
// 1040 B (16B-aligned rows; b128 loads are 2-way bank limited, the floor).
#define BSTRIDE   520
#define SBT_BYTES (RANK * BSTRIDE * 2)            // 66560 B per buffer
#define CHUNKS    (SBT_BYTES / 16)                // 4160 b128 transfers

// Per-wave v buffer, f16, padded rows (144 B: 16B-aligned, bank-spread).
#define VSTRIDE    72
#define VBUF_BYTES (PTS_W * VSTRIDE * 2)          // 2304 B per wave
#define VBUF_TOTAL (WAVES * VBUF_BYTES)           // 18432 B (16B-aligned)

// v-buffers first (DS imm-offset range), then the two B buffers.
#define SBT0_OFF  VBUF_TOTAL
#define SBT1_OFF  (VBUF_TOTAL + SBT_BYTES)
#define LDS_BYTES (VBUF_TOTAL + 2 * SBT_BYTES)    // 151552 B

// Overlays inside buffer 0 (only touched after the last level's barrier):
#define PSTRIDE   72                              // payload^T row stride (f16)
#define OBUF_OFF  (SBT0_OFF + 8192)               // per-wave 16x32 f32 out tiles

union AB16 { unsigned int u[8]; v4u q[2]; v16h h; };

__device__ __forceinline__ unsigned int pkf16(float a, float b) {
  unsigned int lo = (unsigned int)__builtin_bit_cast(unsigned short, (_Float16)a);
  unsigned int hi = (unsigned int)__builtin_bit_cast(unsigned short, (_Float16)b);
  return (hi << 16) | lo;
}

__device__ __forceinline__ int quant8(float c) {
  float f = floorf((c * 0.5f + 0.5f) * 256.0f);
  f = fminf(255.0f, fmaxf(0.0f, f));
  return (int)f;
}

__device__ __forceinline__ unsigned int lds_off(const void* p) {
  // gfx1250 flat->LDS mapping keeps the LDS byte offset in the low 32 bits.
  return (unsigned int)(uintptr_t)p;
}

// Async DMA one full sBT image (SBT_BYTES) from global into LDS.
__device__ __forceinline__ void async_fill(unsigned int lds_base,
                                           const _Float16* src, int tid) {
  unsigned long long sbase = (unsigned long long)(uintptr_t)src;
  for (int i = 0; i < (CHUNKS + 255) / 256; ++i) {
    const int chunk = i * 256 + tid;
    if (chunk < CHUNKS) {
      const unsigned int voff  = (unsigned int)chunk * 16u;
      const unsigned int laddr = lds_base + voff;
      asm volatile("global_load_async_to_lds_b128 %0, %1, %2 offset:0"
                   :: "v"(laddr), "v"(voff), "s"(sbase)
                   : "memory");
    }
  }
}

// Synchronous fallback fill: convert + transpose cores[l] (f32) into LDS f16.
__device__ __forceinline__ void sync_fill(_Float16* sbt, const float* cg, int tid) {
#pragma unroll 4
  for (int i = 0; i < (KDIM * RANK) / 256; ++i) {
    const int flat = i * 256 + tid;                 // = k*64 + s
    sbt[(flat & 63) * BSTRIDE + (flat >> 6)] = (_Float16)cg[flat];
  }
}

// Pre-pass: cores (7,512,64) f32 -> per-level padded LDS image (f16, s-major).
__global__ __launch_bounds__(256) void convert_cores_kernel(
    const float* __restrict__ cores, _Float16* __restrict__ wc) {
  const int g   = blockIdx.x * 256 + threadIdx.x;   // 0 .. 7*32768-1
  const int l   = g >> 15;
  const int rem = g & 32767;                        // = k*64 + s
  const int k = rem >> 6, s = rem & 63;
  wc[(size_t)l * (RANK * BSTRIDE) + s * BSTRIDE + k] = (_Float16)cores[g];
}

template <bool ASYNC>
__global__ __launch_bounds__(256, 1) void qtt_radiance_kernel(
    const float* __restrict__ coords,    // (NTOT, 3)
    const float* __restrict__ viewdirs,  // (NB, 3)
    const float* __restrict__ core0,     // (8, 64)
    const float* __restrict__ cores,     // (7, 512, 64) f32
    const float* __restrict__ payload,   // (64, 28)
    const _Float16* __restrict__ wcores, // pre-converted images (ASYNC only)
    float* __restrict__ out)             // rgb (NTOT*3) then sigma (NTOT)
{
  __shared__ char lds[LDS_BYTES];
  unsigned short* vbufAll = (unsigned short*)lds;   // offset 0: imm DS offsets
  char* sbt0 = lds + SBT0_OFF;
  char* sbt1 = lds + SBT1_OFF;
  _Float16* pbuf    = (_Float16*)sbt0;              // 32 rows x PSTRIDE f16
  float*    obufAll = (float*)(lds + OBUF_OFF);     // per-wave 16x32 f32

  const int tid    = threadIdx.x;
  const int wave   = tid >> 5;
  const int lane   = tid & 31;
  const int lanelo = lane & 15;        // point (M) / column (N) sub-index
  const int hi     = lane >> 4;        // which K half-group this lane feeds

  if (ASYNC) async_fill(lds_off(sbt0), wcores, tid);  // level 1 DMA kick-off

  const int n_pt = blockIdx.x * PTS_WG + wave * PTS_W + lanelo;

  // ---- per-point digits + inside mask (lanes L and L+16 serve point L%16) --
  const float cx = coords[n_pt * 3 + 0];
  const float cy = coords[n_pt * 3 + 1];
  const float cz = coords[n_pt * 3 + 2];
  const float inside =
      (fabsf(cx) <= 1.0f && fabsf(cy) <= 1.0f && fabsf(cz) <= 1.0f) ? 1.0f : 0.0f;
  const int ix = quant8(cx), iy = quant8(cy), iz = quant8(cz);
  unsigned int dpack = 0;
#pragma unroll
  for (int l = 0; l < NLEV; ++l) {
    const int s = 7 - l;
    dpack |= (unsigned int)((((ix >> s) & 1) << 2) | (((iy >> s) & 1) << 1) |
                            ((iz >> s) & 1)) << (3 * l);
  }

  // ---- level 0: v = core0[digit0] into this wave's f16 v-buffer ------------
  unsigned short* vbuf = vbufAll + wave * (PTS_W * VSTRIDE);
  {
    const int d0 = (int)(dpack & 7u);
    const float* src = core0 + d0 * RANK + hi * 32;
    unsigned int* dst = (unsigned int*)(vbuf + lanelo * VSTRIDE + hi * 32);
#pragma unroll
    for (int j = 0; j < 16; ++j) dst[j] = pkf16(src[2 * j], src[2 * j + 1]);
  }

  const v8f zero8 = {0.f, 0.f, 0.f, 0.f, 0.f, 0.f, 0.f, 0.f};

  // ---- levels 1..7: expanded-A WMMA GEMM, double-buffered B in LDS ---------
  for (int l = 1; l < NLEV; ++l) {
    _Float16* cur;
    if (ASYNC) {
      cur = (_Float16*)(((l - 1) & 1) ? sbt1 : sbt0);
      asm volatile("s_wait_asynccnt 0x0" ::: "memory");  // my DMA of cur done
      __syncthreads();  // everyone's DMA done; everyone off the nxt buffer
      if (l < NLEV - 1)
        async_fill(lds_off((l & 1) ? sbt1 : sbt0),
                   wcores + (size_t)l * (RANK * BSTRIDE), tid);
    } else {
      cur = (_Float16*)sbt0;
      __syncthreads();
      sync_fill(cur, cores + (size_t)(l - 1) * KDIM * RANK, tid);
      __syncthreads();
    }

    const unsigned int d    = (dpack >> (3 * l)) & 7u;
    const unsigned int slot = d >> 1;           // VGPR within 4
    const unsigned int shft = (d & 1u) << 4;    // low/high f16 half

    v8f acc[4];
#pragma unroll
    for (int t = 0; t < 4; ++t) acc[t] = zero8;

    const unsigned short* vrow = vbuf + lanelo * VSTRIDE;
#pragma unroll 2
    for (int c = 0; c < 16; ++c) {              // K = 512 in chunks of 32
      // expanded A operand: one nonzero per 8-element half (f16 bits direct)
      const unsigned int hlo = ((unsigned int)vrow[4 * c + 2 * hi]) << shft;
      const unsigned int hhi = ((unsigned int)vrow[4 * c + 2 * hi + 1]) << shft;
      AB16 aa;
#pragma unroll
      for (int j = 0; j < 4; ++j) {
        aa.u[j]     = (slot == (unsigned)j) ? hlo : 0u;
        aa.u[4 + j] = (slot == (unsigned)j) ? hhi : 0u;
      }
      const v16h amat = aa.h;

      const int kbase = 32 * c + 16 * hi;
#pragma unroll
      for (int t = 0; t < 4; ++t) {
        const _Float16* bp = cur + (16 * t + lanelo) * BSTRIDE + kbase;
        AB16 bb;
        bb.q[0] = *(const v4u*)(bp);
        bb.q[1] = *(const v4u*)(bp + 8);
        acc[t] = __builtin_amdgcn_wmma_f32_16x16x32_f16(
            false, amat, false, bb.h, (short)0, acc[t], false, false);
      }
    }

    // D layout: VGPR i, lane L -> row p = i + 8*(L/16), col N = 16t + L%16
#pragma unroll
    for (int t = 0; t < 4; ++t)
#pragma unroll
      for (int i = 0; i < 8; ++i)
        vbuf[(i + 8 * hi) * VSTRIDE + 16 * t + lanelo] =
            __builtin_bit_cast(unsigned short, (_Float16)acc[t][i]);
  }

  // ---- payload stage: out = v @ payload (64x28, padded to 64x32) -----------
  __syncthreads();  // level-7 read sbt0; done before overlaying pbuf
  for (int i = tid; i < 32 * PSTRIDE; i += 256) pbuf[i] = (_Float16)0.0f;
  __syncthreads();
  for (int i = tid; i < RANK * PAY; i += 256) {
    const int r = i / PAY;
    const int j = i - r * PAY;
    pbuf[j * PSTRIDE + r] = (_Float16)payload[i];  // payload^T, s-major
  }
  __syncthreads();

  v8f oacc[2];
  oacc[0] = zero8;
  oacc[1] = zero8;
  {
    const unsigned short* vrow = vbuf + lanelo * VSTRIDE;
#pragma unroll
    for (int c = 0; c < 2; ++c) {               // K = 64 in chunks of 32
      AB16 aa;
      aa.q[0] = *(const v4u*)(vrow + 32 * c + 16 * hi);
      aa.q[1] = *(const v4u*)(vrow + 32 * c + 16 * hi + 8);
      const v16h amat = aa.h;
#pragma unroll
      for (int t = 0; t < 2; ++t) {
        const _Float16* bp = pbuf + (16 * t + lanelo) * PSTRIDE + 32 * c + 16 * hi;
        AB16 bb;
        bb.q[0] = *(const v4u*)(bp);
        bb.q[1] = *(const v4u*)(bp + 8);
        oacc[t] = __builtin_amdgcn_wmma_f32_16x16x32_f16(
            false, amat, false, bb.h, (short)0, oacc[t], false, false);
      }
    }
  }

  // ---- stash out tiles, then SH reduction + mask + global store ------------
  float* obuf = obufAll + wave * (PTS_W * 32);
#pragma unroll
  for (int t = 0; t < 2; ++t)
#pragma unroll
    for (int i = 0; i < 8; ++i)
      obuf[(i + 8 * hi) * 32 + 16 * t + lanelo] = oacc[t][i];
  // obuf is per-wave; same-wave LDS ordering covers the reads below.

  if (lane < 16) {
    const int n = n_pt;
    const int b = n >> 8;  // R = 256 rays per batch entry
    const float x = viewdirs[b * 3 + 0];
    const float y = viewdirs[b * 3 + 1];
    const float z = viewdirs[b * 3 + 2];
    const float C1 = 0.4886025119029199f;
    float sh[9];
    sh[0] = 0.28209479177387814f;
    sh[1] = -C1 * y;
    sh[2] = C1 * z;
    sh[3] = -C1 * x;
    sh[4] = 1.0925484305920792f * x * y;
    sh[5] = -1.0925484305920792f * y * z;
    sh[6] = 0.31539156525252005f * (2.0f * z * z - x * x - y * y);
    sh[7] = -1.0925484305920792f * x * z;
    sh[8] = 0.5462742152960396f * (x * x - y * y);

    const float* ob = obuf + lane * 32;
    float r0 = 0.f, r1 = 0.f, r2 = 0.f;
#pragma unroll
    for (int m = 0; m < 9; ++m) {
      r0 += ob[m] * sh[m];
      r1 += ob[9 + m] * sh[m];
      r2 += ob[18 + m] * sh[m];
    }
    out[n * 3 + 0] = r0 * inside;
    out[n * 3 + 1] = r1 * inside;
    out[n * 3 + 2] = r2 * inside;
    out[NTOT * 3 + n] = ob[27] * inside;
  }
}

extern "C" void kernel_launch(void* const* d_in, const int* in_sizes, int n_in,
                              void* d_out, int out_size, void* d_ws, size_t ws_size,
                              hipStream_t stream) {
  (void)in_sizes; (void)n_in; (void)out_size;
  const float* coords   = (const float*)d_in[0];
  const float* viewdirs = (const float*)d_in[1];
  const float* core0    = (const float*)d_in[2];
  const float* cores    = (const float*)d_in[3];
  const float* payload  = (const float*)d_in[4];
  float* out = (float*)d_out;

  const size_t need = (size_t)7 * RANK * BSTRIDE * sizeof(unsigned short);
  const dim3 grid(NTOT / PTS_WG);  // 2048 workgroups of 256 threads (8 waves)

  if (ws_size >= need) {
    _Float16* wcores = (_Float16*)d_ws;
    convert_cores_kernel<<<dim3((7 * KDIM * RANK) / 256), dim3(256), 0, stream>>>(
        cores, wcores);
    qtt_radiance_kernel<true><<<grid, dim3(256), 0, stream>>>(
        coords, viewdirs, core0, cores, payload, wcores, out);
  } else {
    qtt_radiance_kernel<false><<<grid, dim3(256), 0, stream>>>(
        coords, viewdirs, core0, cores, payload, nullptr, out);
  }
}
